// CBOWHierarchicalSoftmax_90254442758231
// MI455X (gfx1250) — matrix operand order repaired and checked by reading, not compiled, not run on hardware
//
#include <hip/hip_runtime.h>

#define BATCH   16384
#define CTXW    10
#define MAXC    24
#define EMBED   128

typedef __attribute__((ext_vector_type(16))) __bf16 v16bf;
typedef __attribute__((ext_vector_type(8)))  float  v8f;

union BF16x16 { v16bf v; unsigned u[8]; };

// Pack two f32 into a bf16 pair (truncation) with a single v_perm_b32:
// result = { hi16(y), hi16(x) }.  Selector 0x07060302: result bytes
// {S1.b2, S1.b3, S0.b2, S0.b3} with S1=x, S0=y.
__device__ __forceinline__ unsigned pack_bf16_trunc(float x, float y) {
    return __builtin_amdgcn_perm(__float_as_uint(y), __float_as_uint(x), 0x07060302u);
}

__global__ void cbow_zero_out(float* out) {
    if (threadIdx.x == 0 && blockIdx.x == 0) out[0] = 0.0f;
}

// One wave (32 lanes) handles a tile of 16 batches.
// A operand: 16x32 bf16 context-vector slice, built in-register per the CDNA5
//            16-bit A layout (lane<16: M=lane, K={0..7,16..23}; lane>=16:
//            M=lane-16, K={8..15,24..31} per 32-K chunk).
// B operand: 32x16 bf16 node-embedding slice (lane%16 = column N = batch-in-tile,
//            lane/16 selects K half).
// C = A*B via 4 chained v_wmma_f32_16x16x32_bf16 over K=128; only the diagonal
// C[n][n] = ctx[n] . node[n] is consumed.
__global__ __launch_bounds__(256)
void cbow_hs_kernel(const int*           __restrict__ cw,      // [B, CTXW]
                    const int*           __restrict__ paths,   // [B, MAXC]
                    const int*           __restrict__ codes,   // [B, MAXC]
                    const unsigned char* __restrict__ mask,    // [B, MAXC] bool
                    const float*         __restrict__ Win,     // [VOCAB, EMBED]
                    const float*         __restrict__ Wint,    // [VOCAB-1, EMBED]
                    float*               __restrict__ out)
{
    const int  lane = threadIdx.x & 31;
    const int  wave = threadIdx.x >> 5;
    const int  tile = blockIdx.x * 8 + wave;      // 16 batches per tile
    const int  half = lane >> 4;                  // K-half owned by this lane
    const int  mrow = lane & 15;                  // A-row / B-column owned
    const long brow = (long)tile * 16 + mrow;

    // ---- Step 1: CBOW context vector (mean over CTXW embeddings), accumulated
    //      in f32 directly in A-operand element order for this lane.
    float acc[64];
    #pragma unroll
    for (int i = 0; i < 64; ++i) acc[i] = 0.0f;

    #pragma unroll
    for (int c = 0; c < CTXW; ++c) {
        const int w = cw[brow * CTXW + c];
        const float4* row = reinterpret_cast<const float4*>(Win + (long)w * EMBED);
        #pragma unroll
        for (int q = 0; q < 4; ++q) {            // 32-K chunk
            #pragma unroll
            for (int t = 0; t < 2; ++t) {        // the two 8-elem groups in a chunk
                const int d0 = q * 32 + t * 16 + half * 8;  // multiple of 8
                const float4 u0 = row[d0 / 4];
                const float4 u1 = row[d0 / 4 + 1];
                const int g = (q * 2 + t) * 8;
                acc[g+0] += u0.x; acc[g+1] += u0.y; acc[g+2] += u0.z; acc[g+3] += u0.w;
                acc[g+4] += u1.x; acc[g+5] += u1.y; acc[g+6] += u1.z; acc[g+7] += u1.w;
            }
        }
    }

    BF16x16 A[4];
    #pragma unroll
    for (int q = 0; q < 4; ++q)
        #pragma unroll
        for (int e = 0; e < 8; ++e)
            A[q].u[e] = pack_bf16_trunc(acc[q * 16 + 2 * e]     * 0.1f,
                                        acc[q * 16 + 2 * e + 1] * 0.1f); // mean/10

    // ---- Step 2: hierarchical-softmax path scores + loss
    const int  r      = half ? (lane - 24) : lane;     // diag element index (active lanes)
    const bool active = (lane < 8) || (lane >= 24);
    const long bn     = (long)tile * 16 + (lane & 15); // batch whose diag this lane owns

    float lsum = 0.0f;

    for (int l = 0; l < MAXC; ++l) {
        const int p = paths[brow * MAXC + l];
        const float4* nrow = reinterpret_cast<const float4*>(Wint + (long)p * EMBED);

        v8f c8 = {};
        #pragma unroll
        for (int q = 0; q < 4; ++q) {
            const int d0 = q * 32 + half * 8;          // multiple of 8
            const float4 u0 = nrow[d0 / 4 + 0];
            const float4 u1 = nrow[d0 / 4 + 1];
            const float4 u2 = nrow[d0 / 4 + 4];
            const float4 u3 = nrow[d0 / 4 + 5];
            BF16x16 Bq;
            Bq.u[0] = pack_bf16_trunc(u0.x, u0.y);
            Bq.u[1] = pack_bf16_trunc(u0.z, u0.w);
            Bq.u[2] = pack_bf16_trunc(u1.x, u1.y);
            Bq.u[3] = pack_bf16_trunc(u1.z, u1.w);
            Bq.u[4] = pack_bf16_trunc(u2.x, u2.y);
            Bq.u[5] = pack_bf16_trunc(u2.z, u2.w);
            Bq.u[6] = pack_bf16_trunc(u3.x, u3.y);
            Bq.u[7] = pack_bf16_trunc(u3.z, u3.w);
            // (neg_a, A, neg_b, B, c_mod, C, reuse_a, reuse_b)
            c8 = __builtin_amdgcn_wmma_f32_16x16x32_bf16(
                     false, A[q].v, false, Bq.v, (short)0, c8, false, false);
        }

        // Pull this lane's diagonal score out of the accumulator tile.
        float sc;
        switch (r & 7) {
            case 0: sc = c8[0]; break;
            case 1: sc = c8[1]; break;
            case 2: sc = c8[2]; break;
            case 3: sc = c8[3]; break;
            case 4: sc = c8[4]; break;
            case 5: sc = c8[5]; break;
            case 6: sc = c8[6]; break;
            default: sc = c8[7]; break;
        }

        if (active) {
            const int           code = codes[bn * MAXC + l];
            const unsigned char mk   = mask [bn * MAXC + l];
            const float t = code ? sc : -sc;               // sign = 2*code-1
            // -log_sigmoid(t) = softplus(-t), numerically stable form:
            const float loss = fmaxf(-t, 0.0f) + log1pf(__expf(-fabsf(t)));
            if (mk) lsum += loss;
        }
    }

    // ---- Wave reduction + global accumulation (scaled by 1/BATCH)
    #pragma unroll
    for (int off = 16; off > 0; off >>= 1)
        lsum += __shfl_down(lsum, off, 32);
    if (lane == 0)
        atomicAdd(out, lsum * (1.0f / (float)BATCH));
}

extern "C" void kernel_launch(void* const* d_in, const int* in_sizes, int n_in,
                              void* d_out, int out_size, void* d_ws, size_t ws_size,
                              hipStream_t stream) {
    const int*           cw    = (const int*)          d_in[0];
    const int*           paths = (const int*)          d_in[1];
    const int*           codes = (const int*)          d_in[2];
    const unsigned char* mask  = (const unsigned char*)d_in[3];
    const float*         Win   = (const float*)        d_in[4];
    const float*         Wint  = (const float*)        d_in[5];
    float*               out   = (float*)              d_out;

    cbow_zero_out<<<1, 64, 0, stream>>>(out);

    const int tiles  = BATCH / 16;          // 1024 waves
    const int blocks = tiles / 8;           // 8 waves (256 threads) per block
    cbow_hs_kernel<<<blocks, 256, 0, stream>>>(cw, paths, codes, mask, Win, Wint, out);
}
